// SelfAttnV3_12773232738987
// MI455X (gfx1250) — matrix operand order repaired and checked by reading, not compiled, not run on hardware
//
#include <hip/hip_runtime.h>

// ---------------------------------------------------------------------------
// CDNA5 (gfx1250, wave32) bf16-WMMA self-attention pipeline.
//   Q/K/V = X @ W.T + b            (bf16 in, fp32 acc, bf16 out)
//   P     = softmax(mask(Q K^T/32))(bf16)
//   h     = P @ V                  (bf16)
//   out   = h @ wo.T + bo          (fp32)
// Intermediates (96MB) fit in the 192MB L2, so the staged pipeline does not
// round-trip HBM between kernels.
//
// GEMM: 256 threads (8 waves), 128x128 block tile, 32x64 wave tile,
// K-step 32, double-buffered LDS (one barrier per K-step), staging split
// into load-phase / store-phase for memory-level parallelism.
// ---------------------------------------------------------------------------

typedef __attribute__((ext_vector_type(16))) __bf16 v16bf;
typedef __attribute__((ext_vector_type(8)))  __bf16 v8bf;
typedef __attribute__((ext_vector_type(4)))  __bf16 v4bf;
typedef __attribute__((ext_vector_type(8)))  float  v8f;
typedef __attribute__((ext_vector_type(4)))  float  v4f;

#define LDST 40  // LDS row stride in bf16 elements (80 bytes: 16B-aligned, bank-spread)

template <typename T> struct Vec4;
template <> struct Vec4<float>  { typedef v4f  type; };
template <> struct Vec4<__bf16> { typedef v4bf type; };

__device__ inline v4bf to_bf4(v4f f) {
    v4bf r;
    r.x = (__bf16)f.x; r.y = (__bf16)f.y; r.z = (__bf16)f.z; r.w = (__bf16)f.w;
    return r;
}
__device__ inline v4bf to_bf4(v4bf f) { return f; }

union FragU {
    v16bf v;
    v8bf  h[2];
};

// A fragment (16x32 bf16, M x K), layout per ISA 7.12.2:
// lanes 0-15: M=lane, K = {0..7, 16..23}; lanes 16-31: M=lane-16, K = {8..15, 24..31}
__device__ inline v16bf load_frag_A(const __bf16* tile, int lane) {
    const int m  = lane & 15;
    const int kh = (lane >> 4) * 8;
    const __bf16* p = tile + m * LDST + kh;
    FragU f;
    f.h[0] = *(const v8bf*)(p);        // K = kh .. kh+7
    f.h[1] = *(const v8bf*)(p + 16);   // K = kh+16 .. kh+23
    return f.v;
}

// B fragment (32x16 bf16, K x N), tile stored N-major in LDS ([n][k]):
// lanes 0-15: N=lane, K = {0..7, 16..23}; lanes 16-31: N=lane-16, K = {8..15, 24..31}
__device__ inline v16bf load_frag_B(const __bf16* tile, int lane) {
    const int n  = lane & 15;
    const int kb = (lane >> 4) * 8;
    const __bf16* p = tile + n * LDST + kb;
    FragU f;
    f.h[0] = *(const v8bf*)(p);
    f.h[1] = *(const v8bf*)(p + 16);
    return f.v;
}

// C[m,n] = scale * sum_k A[m,k]*B'[k,n] (+ bias[n])
// B_IS_NK=true : B given as [N,K] row-major (weights, K-matrix)  -> row copy
// B_IS_NK=false: B given as [K,N] row-major (V matrix)           -> transpose in LDS
// Requires M,N % 128 == 0, K % 32 == 0.
template <typename AT, typename BT, bool B_IS_NK, typename OT, bool HAS_BIAS>
__global__ __launch_bounds__(256)
void gemm_wmma_kernel(const AT* __restrict__ Ag, const BT* __restrict__ Bg,
                      const float* __restrict__ bias, OT* __restrict__ Cg,
                      int M, int N, int K, float scale,
                      long long sA, long long sB, long long sC)
{
    __shared__ __attribute__((aligned(16))) __bf16 As[2][128 * LDST];
    __shared__ __attribute__((aligned(16))) __bf16 Bs[2][128 * LDST];

    const int z = blockIdx.z;
    const AT* A = Ag + (size_t)z * sA;
    const BT* B = Bg + (size_t)z * sB;
    OT*       C = Cg + (size_t)z * sC;

    const int n0   = blockIdx.x * 128;
    const int m0   = blockIdx.y * 128;
    const int tid  = threadIdx.x;
    const int lane = tid & 31;
    const int wave = tid >> 5;
    const int wm   = (wave & 3) * 32;   // wave row offset (4 wave-rows x 32)
    const int wn   = (wave >> 2) * 64;  // wave col offset (2 wave-cols x 64)

    // staging index precompute (A / B-NK: 128 rows x 8 vec4; B-KN: 32 k x 32 nvec4)
    typename Vec4<AT>::type ta[4];
    typename Vec4<BT>::type tb[4];

    v8f acc[2][4] = {};

    // ---- prologue: stage K-step 0 into buffer 0 ----
    #pragma unroll
    for (int i = 0; i < 4; ++i) {
        int l = tid + i * 256, r = l >> 3, c = (l & 7) * 4;
        ta[i] = *(const typename Vec4<AT>::type*)(&A[(size_t)(m0 + r) * K + c]);
    }
    if (B_IS_NK) {
        #pragma unroll
        for (int i = 0; i < 4; ++i) {
            int l = tid + i * 256, r = l >> 3, c = (l & 7) * 4;
            tb[i] = *(const typename Vec4<BT>::type*)(&B[(size_t)(n0 + r) * K + c]);
        }
    } else {
        #pragma unroll
        for (int i = 0; i < 4; ++i) {
            int l = tid + i * 256, kr = l >> 5, nc = (l & 31) * 4;
            tb[i] = *(const typename Vec4<BT>::type*)(&B[(size_t)kr * N + n0 + nc]);
        }
    }
    #pragma unroll
    for (int i = 0; i < 4; ++i) {
        int l = tid + i * 256, r = l >> 3, c = (l & 7) * 4;
        *(v4bf*)(&As[0][r * LDST + c]) = to_bf4(ta[i]);
    }
    if (B_IS_NK) {
        #pragma unroll
        for (int i = 0; i < 4; ++i) {
            int l = tid + i * 256, r = l >> 3, c = (l & 7) * 4;
            *(v4bf*)(&Bs[0][r * LDST + c]) = to_bf4(tb[i]);
        }
    } else {
        #pragma unroll
        for (int i = 0; i < 4; ++i) {
            int l = tid + i * 256, kr = l >> 5, nc = (l & 31) * 4;
            v4bf d = to_bf4(tb[i]);
            Bs[0][(nc + 0) * LDST + kr] = d.x;
            Bs[0][(nc + 1) * LDST + kr] = d.y;
            Bs[0][(nc + 2) * LDST + kr] = d.z;
            Bs[0][(nc + 3) * LDST + kr] = d.w;
        }
    }
    __syncthreads();

    int cur = 0;
    for (int s = 0;; ++s) {
        const int  k_next   = (s + 1) * 32;
        const bool has_next = k_next < K;

        // ---- issue next tile's global loads (no waits in between) ----
        if (has_next) {
            #pragma unroll
            for (int i = 0; i < 4; ++i) {
                int l = tid + i * 256, r = l >> 3, c = (l & 7) * 4;
                ta[i] = *(const typename Vec4<AT>::type*)(&A[(size_t)(m0 + r) * K + k_next + c]);
            }
            if (B_IS_NK) {
                #pragma unroll
                for (int i = 0; i < 4; ++i) {
                    int l = tid + i * 256, r = l >> 3, c = (l & 7) * 4;
                    tb[i] = *(const typename Vec4<BT>::type*)(&B[(size_t)(n0 + r) * K + k_next + c]);
                }
            } else {
                #pragma unroll
                for (int i = 0; i < 4; ++i) {
                    int l = tid + i * 256, kr = l >> 5, nc = (l & 31) * 4;
                    tb[i] = *(const typename Vec4<BT>::type*)(&B[(size_t)(k_next + kr) * N + n0 + nc]);
                }
            }
            if (k_next + 32 < K) {  // prefetch 2 K-steps ahead (global_prefetch_b8)
                int r = tid >> 1;
                __builtin_prefetch(&A[(size_t)(m0 + r) * K + k_next + 32], 0, 0);
            }
        }

        // ---- compute from current buffer: 8 WMMAs per wave ----
        const __bf16* At = As[cur];
        const __bf16* Bt = Bs[cur];
        v16bf a0 = load_frag_A(At + (wm + 0)  * LDST, lane);
        v16bf a1 = load_frag_A(At + (wm + 16) * LDST, lane);
        #pragma unroll
        for (int j = 0; j < 4; ++j) {
            v16bf b = load_frag_B(Bt + (wn + 16 * j) * LDST, lane);
            acc[0][j] = __builtin_amdgcn_wmma_f32_16x16x32_bf16(false, a0, false, b,
                                                                (short)0, acc[0][j], false, false);
            acc[1][j] = __builtin_amdgcn_wmma_f32_16x16x32_bf16(false, a1, false, b,
                                                                (short)0, acc[1][j], false, false);
        }

        if (!has_next) break;

        // ---- store next tile into alternate buffer ----
        const int nxt = cur ^ 1;
        #pragma unroll
        for (int i = 0; i < 4; ++i) {
            int l = tid + i * 256, r = l >> 3, c = (l & 7) * 4;
            *(v4bf*)(&As[nxt][r * LDST + c]) = to_bf4(ta[i]);
        }
        if (B_IS_NK) {
            #pragma unroll
            for (int i = 0; i < 4; ++i) {
                int l = tid + i * 256, r = l >> 3, c = (l & 7) * 4;
                *(v4bf*)(&Bs[nxt][r * LDST + c]) = to_bf4(tb[i]);
            }
        } else {
            #pragma unroll
            for (int i = 0; i < 4; ++i) {
                int l = tid + i * 256, kr = l >> 5, nc = (l & 31) * 4;
                v4bf d = to_bf4(tb[i]);
                Bs[nxt][(nc + 0) * LDST + kr] = d.x;
                Bs[nxt][(nc + 1) * LDST + kr] = d.y;
                Bs[nxt][(nc + 2) * LDST + kr] = d.z;
                Bs[nxt][(nc + 3) * LDST + kr] = d.w;
            }
        }
        __syncthreads();
        cur = nxt;
    }

    // ---- epilogue: C/D layout per ISA 7.12.2 (VGPR v: M=v(+8), N=lane&15) ----
    const int cn    = lane & 15;
    const int rbase = (lane >> 4) * 8;
    #pragma unroll
    for (int i = 0; i < 2; ++i) {
        #pragma unroll
        for (int j = 0; j < 4; ++j) {
            const int gm0 = m0 + wm + i * 16 + rbase;
            const int gn  = n0 + wn + j * 16 + cn;
            const float bv = HAS_BIAS ? bias[gn] : 0.0f;
            #pragma unroll
            for (int v = 0; v < 8; ++v) {
                float r = acc[i][j][v] * scale + bv;
                C[(size_t)(gm0 + v) * N + gn] = (OT)r;
            }
        }
    }
}

// Row softmax with mask (mask==0 -> -inf), in place over bf16 scores.
// One 256-thread block per row, S == 2048, thread owns 8 contiguous elements.
__global__ __launch_bounds__(256)
void softmax_mask_kernel(__bf16* __restrict__ P, const int* __restrict__ mask, int S)
{
    __shared__ float red[8];
    const size_t row = blockIdx.x;
    __bf16*    prow = P + row * (size_t)S;
    const int* mrow = mask + row * (size_t)S;
    const int tid  = threadIdx.x;
    const int lane = tid & 31;
    const int wave = tid >> 5;
    const int base = tid * 8;

    v8bf pv = *(const v8bf*)(&prow[base]);
    int4 m0 = *(const int4*)(&mrow[base]);
    int4 m1 = *(const int4*)(&mrow[base + 4]);
    int  mm[8] = { m0.x, m0.y, m0.z, m0.w, m1.x, m1.y, m1.z, m1.w };

    float vals[8];
    float mx = -3.4e38f;
    #pragma unroll
    for (int i = 0; i < 8; ++i) {
        float v = (float)pv[i];
        if (mm[i] == 0) v = -3.4e38f;
        vals[i] = v;
        mx = fmaxf(mx, v);
    }
    #pragma unroll
    for (int o = 16; o > 0; o >>= 1) mx = fmaxf(mx, __shfl_xor(mx, o, 32));
    if (lane == 0) red[wave] = mx;
    __syncthreads();
    mx = red[0];
    #pragma unroll
    for (int w = 1; w < 8; ++w) mx = fmaxf(mx, red[w]);

    float s = 0.0f;
    #pragma unroll
    for (int i = 0; i < 8; ++i) {
        float e = __expf(vals[i] - mx);
        vals[i] = e;
        s += e;
    }
    #pragma unroll
    for (int o = 16; o > 0; o >>= 1) s += __shfl_xor(s, o, 32);
    __syncthreads();                 // done reading red (max phase)
    if (lane == 0) red[wave] = s;
    __syncthreads();
    s = 0.0f;
    #pragma unroll
    for (int w = 0; w < 8; ++w) s += red[w];
    const float inv = 1.0f / s;

    v8bf outv;
    #pragma unroll
    for (int i = 0; i < 8; ++i) outv[i] = (__bf16)(vals[i] * inv);
    *(v8bf*)(&prow[base]) = outv;
}

extern "C" void kernel_launch(void* const* d_in, const int* in_sizes, int n_in,
                              void* d_out, int out_size, void* d_ws, size_t ws_size,
                              hipStream_t stream)
{
    (void)in_sizes; (void)n_in; (void)out_size; (void)ws_size;

    const float* X    = (const float*)d_in[0];
    const int*   mask = (const int*)  d_in[1];
    const float* wq   = (const float*)d_in[2];
    const float* bq   = (const float*)d_in[3];
    const float* wk   = (const float*)d_in[4];
    const float* bk   = (const float*)d_in[5];
    const float* wv   = (const float*)d_in[6];
    const float* bv   = (const float*)d_in[7];
    const float* wo   = (const float*)d_in[8];
    const float* bo   = (const float*)d_in[9];
    float* out = (float*)d_out;

    const int Bn = 4, S = 2048, H = 1024;
    const int M  = Bn * S;                       // 8192

    // workspace layout (bf16): Q,K,V [M,H] + P [B,S,S] + h [M,H]  = 96MB
    __bf16* Q  = (__bf16*)d_ws;
    __bf16* Kb = Q  + (size_t)M * H;
    __bf16* V  = Kb + (size_t)M * H;
    __bf16* P  = V  + (size_t)M * H;
    __bf16* Hh = P  + (size_t)Bn * S * S;

    const dim3 blk(256);

    // --- QKV projections: [8192,1024] x [1024,1024]^T + bias -> bf16 ---
    {
        dim3 g(H / 128, M / 128, 1);
        gemm_wmma_kernel<float, float, true, __bf16, true><<<g, blk, 0, stream>>>(
            X, wq, bq, Q, M, H, H, 1.0f, 0, 0, 0);
        gemm_wmma_kernel<float, float, true, __bf16, true><<<g, blk, 0, stream>>>(
            X, wk, bk, Kb, M, H, H, 1.0f, 0, 0, 0);
        gemm_wmma_kernel<float, float, true, __bf16, true><<<g, blk, 0, stream>>>(
            X, wv, bv, V, M, H, H, 1.0f, 0, 0, 0);
    }

    // --- scores: P[b] = (Q[b] @ K[b]^T) / sqrt(H); K is [N,K]-major ---
    {
        dim3 g(S / 128, S / 128, Bn);
        gemm_wmma_kernel<__bf16, __bf16, true, __bf16, false><<<g, blk, 0, stream>>>(
            Q, Kb, nullptr, P, S, S, H, 1.0f / 32.0f,
            (long long)S * H, (long long)S * H, (long long)S * S);
    }

    // --- masked softmax over each of the B*S rows (length S) ---
    softmax_mask_kernel<<<dim3(Bn * S), 256, 0, stream>>>(P, mask, S);

    // --- h[b] = P[b] @ V[b]; V is [K,N]-major -> transpose-stage path ---
    {
        dim3 g(H / 128, S / 128, Bn);
        gemm_wmma_kernel<__bf16, __bf16, false, __bf16, false><<<g, blk, 0, stream>>>(
            P, V, nullptr, Hh, S, H, S, 1.0f,
            (long long)S * S, (long long)S * H, (long long)S * H);
    }

    // --- out = h @ wo^T + bo (fp32 result) ---
    {
        dim3 g(H / 128, M / 128, 1);
        gemm_wmma_kernel<__bf16, float, true, float, true><<<g, blk, 0, stream>>>(
            Hh, wo, bo, out, M, H, H, 1.0f, 0, 0, 0);
    }
}
